// LABNet_12086037971184
// MI455X (gfx1250) — matrix operand order repaired and checked by reading, not compiled, not run on hardware
//
#include <hip/hip_runtime.h>
#include <hip/hip_bf16.h>
#include <math.h>
#include <stdint.h>

typedef _Float16 half_t;
typedef __attribute__((ext_vector_type(16))) _Float16 v16h;
typedef __attribute__((ext_vector_type(8)))  float    v8f;

#define LRELU_SLOPE 0.2f

__device__ __forceinline__ int refl(int i, int n) {
  if (i < 0) i = -i;
  if (i >= n) i = 2*n - 2 - i;
  return i;
}

// ---------- weight packing: OIHW f32 -> [tap][coutP][cinP] f16 (zero padded) ----------
__global__ void pack_weights(const float* __restrict__ w, half_t* __restrict__ wp,
                             int Cout, int Cin, int K, int coutP, int cinP) {
  int idx = blockIdx.x * blockDim.x + threadIdx.x;
  int total = K * K * coutP * cinP;
  if (idx >= total) return;
  int ci = idx % cinP;
  int t  = idx / cinP;
  int co = t % coutP;
  t /= coutP;
  float v = 0.f;
  if (co < Cout && ci < Cin)
    v = w[((size_t)(co * Cin + ci) * K + (t / K)) * K + (t % K)];
  wp[idx] = (half_t)v;
}

// ---------- implicit-GEMM conv + bias + LeakyReLU, fully specialized ----------
// KSZ: kernel size (1 or 3); CB: 32-channel K blocks; COT: 16-cout tiles.
// One wave computes a 16(cout) x 16(pixel) tile; taps x K-blocks fully unrolled =>
// KSZ*KSZ*CB straight-line v_wmma_f32_16x16x32_f16 per wave. Border reflection is
// handled lazily in a cold path so no index tables stay live (no SGPR spills).
template <int KSZ, int CB, int COT>
__global__ __launch_bounds__(32)
void conv_wmma(const half_t* __restrict__ in, int inC, int incoff, int Cin,
               const half_t* __restrict__ wp, const float* __restrict__ bias,
               half_t* __restrict__ out, int outC, int outcoff, int Cout,
               int Hh, int Ww, int dil) {
  const int lane = threadIdx.x;
  const int x0 = (int)blockIdx.x << 4;
  const int y  = (int)blockIdx.y;
  const int bz = (int)blockIdx.z;
  const int ct = bz % COT;         // compile-time constant divisor
  const int n  = bz / COT;
  const int co0 = ct << 4;
  const int cinP  = CB * 32;
  const int coutP = COT * 16;

  const int am = lane & 15;            // A-matrix row (M) held by this lane
  const int kb = (lane < 16) ? 0 : 8;  // A-matrix K base for this half-wave
  const int kd = kb >> 1;              // in dwords

  // per-lane B row bases (lane = K channel within 32-block)
  const half_t* rowbase[CB];
  bool cvalid[CB];
#pragma unroll
  for (int cb = 0; cb < CB; ++cb) {
    int c = cb * 32 + lane;
    cvalid[cb]  = (c < Cin);
    rowbase[cb] = in + ((size_t)n * inC + incoff + (c < Cin ? c : 0)) * Hh * (size_t)Ww;
  }

  v8f acc = {};
#pragma unroll
  for (int ky = 0; ky < KSZ; ++ky) {
    const int ysrc = (KSZ == 3) ? refl(y + (ky - 1) * dil, Hh) : y;
    const int yrow = ysrc * Ww;
#pragma unroll
    for (int kx = 0; kx < KSZ; ++kx) {
      const int t = ky * KSZ + kx;
      const half_t* wtap = wp + ((size_t)t * coutP + (co0 + am)) * cinP;
      const int xoff = (KSZ == 3) ? (kx - 1) * dil : 0;
      const int xb = x0 + xoff;
      const bool fast = (xb >= 0) && (xb + 15 < Ww);
#pragma unroll
      for (int cb = 0; cb < CB; ++cb) {
        // A fragment: two 16B-aligned 4-dword runs
        union { uint32_t u[8]; v16h h; } ua;
        const uint32_t* ap = (const uint32_t*)(wtap + cb * 32);
#pragma unroll
        for (int j = 0; j < 4; ++j) {
          ua.u[j]     = ap[kd + j];
          ua.u[4 + j] = ap[8 + kd + j];
        }
        // B fragment: lane&15 = N (pixel x), elements = K (channel) in table order
        v16h b;
        {
          const int xs = fast ? (xb + am) : refl(xb + am, Ww);
          const half_t* ip = in + ((size_t)n * inC + incoff) * Hh * (size_t)Ww + yrow + xs;
#pragma unroll
          for (int i = 0; i < 16; ++i) {
            const int c = cb * 32 + ((i < 8) ? (kb + i) : (16 + kb + i - 8));
            b[i] = (c < Cin) ? ip[(size_t)c * Hh * Ww] : (half_t)0.f;
          }
        }
        acc = __builtin_amdgcn_wmma_f32_16x16x32_f16(false, ua.h, false, b,
                                                     (short)0, acc, false, false);
      }
    }
  }

  // C/D layout: VGPR r: lanes0-15 -> M=r; lanes16-31 -> M=8+r; N = lane&15
  int px = x0 + (lane & 15);
  int mbase = (lane < 16) ? 0 : 8;
#pragma unroll
  for (int r = 0; r < 8; ++r) {
    int co = co0 + mbase + r;
    if (co < Cout) {
      float v = acc[r] + bias[co];
      v = (v >= 0.f) ? v : LRELU_SLOPE * v;
      out[(((size_t)n * outC + outcoff + co) * Hh + y) * Ww + px] = (half_t)v;
    }
  }
}

// ---------- elementwise / layout helpers (compile-time extents -> shifts) ----------
template <int HW>
__global__ void f32_to_f16_ch(const float* __restrict__ in, half_t* __restrict__ out,
                              int C, int outC, int coff, int total) {
  int i = blockIdx.x * blockDim.x + threadIdx.x;
  if (i >= total) return;
  int pix = i % HW; int t = i / HW; int c = t % C; int n = t / C;
  out[((size_t)n * outC + coff + c) * HW + pix] = (half_t)in[i];
}

template <int HW>
__global__ void copy_ch_f16(const half_t* __restrict__ in, int inC, int incoff,
                            half_t* __restrict__ out, int outC, int outcoff,
                            int C, int total) {
  int i = blockIdx.x * blockDim.x + threadIdx.x;
  if (i >= total) return;
  int pix = i % HW; int t = i / HW; int c = t % C; int n = t / C;
  out[((size_t)n * outC + outcoff + c) * HW + pix] =
      in[((size_t)n * inC + incoff + c) * HW + pix];
}

__global__ void f16_to_f32_k(const half_t* __restrict__ in, float* __restrict__ out, int total) {
  int i = blockIdx.x * blockDim.x + threadIdx.x;
  if (i < total) out[i] = (float)in[i];
}

__global__ void add_lab_k(const half_t* __restrict__ a, const float* __restrict__ b,
                          half_t* __restrict__ out, int total) {
  int i = blockIdx.x * blockDim.x + threadIdx.x;
  if (i < total) out[i] = (half_t)((float)a[i] + b[i]);
}

// nearest (torch style: src = floor(dst*in/out)); OH/OW/IH/IW compile-time
template <int OH, int OW, int IH, int IW>
__global__ void nearest_f16_k(const half_t* __restrict__ in, int inC, int incoff,
                              half_t* __restrict__ out, int outC, int outcoff,
                              int C, int total) {
  int i = blockIdx.x * blockDim.x + threadIdx.x;
  if (i >= total) return;
  int xo = i % OW; int t = i / OW; int yo = t % OH; t /= OH;
  int c = t % C; int n = t / C;
  int yi = (yo * IH) / OH, xi = (xo * IW) / OW;
  out[(((size_t)n * outC + outcoff + c) * OH + yo) * OW + xo] =
      in[(((size_t)n * inC + incoff + c) * IH + yi) * IW + xi];
}

template <int OH, int OW, int IH, int IW>
__global__ void nearest_mask_k(const float* __restrict__ in, float* __restrict__ out, int total) {
  int i = blockIdx.x * blockDim.x + threadIdx.x;
  if (i >= total) return;
  int xo = i % OW; int t = i / OW; int yo = t % OH; int n = t / OH;
  int yi = (yo * IH) / OH, xi = (xo * IW) / OW;
  out[i] = in[((size_t)n * IH + yi) * IW + xi];
}

// ---------- channel attention (H=W=256 fixed) ----------
__global__ void ca_std_k(const half_t* __restrict__ f, float* __restrict__ sd) {
  const int Hh = 256, Ww = 256;
  __shared__ float rs[256], rq[256];
  int nc = blockIdx.x, tid = threadIdx.x;
  const half_t* p = f + (size_t)nc * Hh * Ww;
  const int cnt = (Hh - 2) * (Ww - 2);
  float s = 0.f, q = 0.f;
  for (int i = tid; i < cnt; i += blockDim.x) {
    int yy = i / (Ww - 2) + 1, xx = i % (Ww - 2) + 1;
    float v = 4.f * (float)p[yy*Ww+xx] - (float)p[(yy-1)*Ww+xx] - (float)p[(yy+1)*Ww+xx]
              - (float)p[yy*Ww+xx-1] - (float)p[yy*Ww+xx+1];
    s += v; q += v * v;
  }
  rs[tid] = s; rq[tid] = q; __syncthreads();
  for (int st = 128; st > 0; st >>= 1) {
    if (tid < st) { rs[tid] += rs[tid+st]; rq[tid] += rq[tid+st]; }
    __syncthreads();
  }
  if (tid == 0) {
    float mean = rs[0] / cnt;
    float var  = (rq[0] - rs[0] * mean) / (cnt - 1);
    sd[nc] = sqrtf(fmaxf(var, 0.f));
  }
}

__global__ void ca_fc_k(const float* __restrict__ sd,
                        const float* __restrict__ w1, const float* __restrict__ b1,
                        const float* __restrict__ w2, const float* __restrict__ b2,
                        float* __restrict__ g, int C, int Ch) {
  __shared__ float h[32];
  int n = blockIdx.x, t = threadIdx.x;
  if (t < Ch) {
    float a = b1[t];
    for (int i = 0; i < C; ++i) a += w1[t * C + i] * sd[n * C + i];
    h[t] = (a >= 0.f) ? a : LRELU_SLOPE * a;
  }
  __syncthreads();
  if (t < C) {
    float a = b2[t];
    for (int i = 0; i < Ch; ++i) a += w2[t * Ch + i] * h[i];
    g[n * C + t] = 1.f / (1.f + __expf(-a));
  }
}

template <int HW>
__global__ void ca_scale_k(half_t* __restrict__ f, const float* __restrict__ g,
                           int C, int total) {
  int i = blockIdx.x * blockDim.x + threadIdx.x;
  if (i >= total) return;
  int c = (i / HW) % C; int n = i / (HW * C);
  f[i] = (half_t)((float)f[i] * g[n * C + c]);
}

// ---------- spatial attention core (48x48 => P=2304 positions) ----------
#define SA_P 2304
__global__ void spa_attn_k(const half_t* __restrict__ q, const half_t* __restrict__ kk,
                           const float* __restrict__ m, const float* __restrict__ md,
                           half_t* __restrict__ kout, int C3) {
  const int P = SA_P;
  int p = blockIdx.x, n = blockIdx.y;
  int tid = threadIdx.x, nt = blockDim.x;
  const half_t* kb = kk + (size_t)n * C3 * P;
  const half_t* qb = q  + (size_t)n * C3 * P;
  half_t* ob = kout + (size_t)n * C3 * P;
  bool ms = (m[n * P + p] == 1.0f);
  if (!ms) {  // non-shadow queries: keys pass through
    for (int c = tid; c < C3; c += nt) ob[(size_t)c * P + p] = kb[(size_t)c * P + p];
    return;
  }
  __shared__ float qv[64];
  __shared__ float sc[SA_P];
  __shared__ float red[256];
  for (int c = tid; c < C3; c += nt) qv[c] = (float)qb[(size_t)c * P + p];
  __syncthreads();
  float scale = sqrtf((float)C3);
  float lmax = -1e30f;
  for (int j = tid; j < P; j += nt) {
    bool ring = ((md[n * P + j] - m[n * P + j]) == 1.0f);
    float s = -1e30f;
    if (ring) {
      float acc = 0.f;
#pragma unroll 8
      for (int c = 0; c < C3; ++c) acc += qv[c] * (float)kb[(size_t)c * P + j];
      s = acc / scale;
    }
    sc[j] = s;
    lmax = fmaxf(lmax, s);
  }
  red[tid] = lmax; __syncthreads();
  for (int st = nt >> 1; st > 0; st >>= 1) {
    if (tid < st) red[tid] = fmaxf(red[tid], red[tid + st]);
    __syncthreads();
  }
  float gmax = red[0]; __syncthreads();
  float lsum = 0.f;
  for (int j = tid; j < P; j += nt) {
    float e = (sc[j] > -1e29f) ? __expf(sc[j] - gmax) : 0.f;
    sc[j] = e; lsum += e;
  }
  red[tid] = lsum; __syncthreads();
  for (int st = nt >> 1; st > 0; st >>= 1) {
    if (tid < st) red[tid] += red[tid + st];
    __syncthreads();
  }
  float denom = red[0];
  float inv = (denom > 0.f) ? 1.f / denom : 0.f;
  __syncthreads();
  for (int c = tid; c < C3; c += nt) {
    const half_t* kr = kb + (size_t)c * P;
    float acc = 0.f;
#pragma unroll 8
    for (int j = 0; j < P; ++j) acc += sc[j] * (float)kr[j];
    ob[(size_t)c * P + p] = (denom > 0.f) ? (half_t)(acc * inv) : kr[p];
  }
}

// ============================ host orchestration ============================
struct CP  { const float* w; const float* b; int cin, cout, k; };
struct CAp { const float *w1, *b1, *w2, *b2; };
struct BBp { CP c[14]; CAp ca; int cout; };
struct SAp { CP s, ns, fin; };

extern "C" void kernel_launch(void* const* d_in, const int* in_sizes, int n_in,
                              void* d_out, int out_size, void* d_ws, size_t ws_size,
                              hipStream_t stream) {
  (void)in_sizes; (void)n_in; (void)out_size; (void)ws_size;
  const int N = 2, H = 256, W = 256, HW = H * W;
  const int DS = 48, PP = DS * DS;           // down_w = down_h = 48 (fixed by harness)
  const size_t PF = (size_t)N * HW;

  const float* lab    = (const float*)d_in[0];
  const float* maskgt = (const float*)d_in[1];
  const float* mask_d = (const float*)d_in[2];
  // d_in[3], d_in[4] = down_w/down_h device scalars (not host-readable under capture)

  // ---- parameter parsing (depth-first dict order) ----
  int cur = 5;
  auto cp = [&](int cin, int cout, int k) {
    CP p; p.w = (const float*)d_in[cur]; p.b = (const float*)d_in[cur + 1];
    p.cin = cin; p.cout = cout; p.k = k; cur += 2; return p;
  };
  auto bb = [&](int cin, int cout) {
    BBp b;
    b.c[0]  = cp(cin, 16, 3); b.c[1]  = cp(cin, 32, 3); b.c[2]  = cp(cin, 48, 3); b.c[3]  = cp(96, 32, 1);
    b.c[4]  = cp(32, 16, 3);  b.c[5]  = cp(32, 32, 3);  b.c[6]  = cp(32, 48, 3);  b.c[7]  = cp(96, 32, 1);
    b.c[8]  = cp(32, 16, 3);  b.c[9]  = cp(32, 32, 3);  b.c[10] = cp(32, 48, 3);  b.c[11] = cp(96, 32, 1);
    b.c[12] = cp(96, 32, 1);  b.c[13] = cp(32, cout, 3);
    b.ca.w1 = (const float*)d_in[cur];     b.ca.b1 = (const float*)d_in[cur + 1];
    b.ca.w2 = (const float*)d_in[cur + 2]; b.ca.b2 = (const float*)d_in[cur + 3];
    cur += 4; b.cout = cout; return b;
  };
  auto sa = [&]() { SAp s; s.s = cp(32, 48, 3); s.ns = cp(32, 48, 3); s.fin = cp(80, 32, 1); return s; };

  CP  convst = cp(3, 32, 3);
  BBp bb1 = bb(33, 32), bb2 = bb(33, 32);
  CP  c12l = cp(64, 32, 1), c12ab = cp(64, 32, 1);
  BBp bb3 = bb(32, 32), bb4 = bb(32, 32);
  CP  c34l = cp(64, 32, 1), c34ab = cp(64, 32, 1);
  SAp sa1 = sa(), sa2 = sa();
  BBp bb5 = bb(32, 32), bb6 = bb(32, 32);
  CP  c56l = cp(64, 32, 1), c56ab = cp(64, 32, 1);
  SAp sa3 = sa(), sa4 = sa();
  BBp bb7 = bb(32, 1), bb8 = bb(32, 2);
  CP  convfin = cp(3, 3, 3);

  // ---- workspace carve ----
  char* wsp = (char*)d_ws;
  auto alloc = [&](size_t bytes) -> void* {
    void* p = (void*)wsp; wsp += (bytes + 255) & ~(size_t)255; return p;
  };
  half_t* bufIn3 = (half_t*)alloc(3  * PF * 2);
  half_t* im     = (half_t*)alloc(33 * PF * 2);
  half_t* t1     = (half_t*)alloc(96 * PF * 2);
  half_t* fcat   = (half_t*)alloc(96 * PF * 2);
  half_t* t2     = (half_t*)alloc(32 * PF * 2);
  half_t* cat12  = (half_t*)alloc(64 * PF * 2);
  half_t* bufL   = (half_t*)alloc(32 * PF * 2);
  half_t* bufAB  = (half_t*)alloc(32 * PF * 2);
  half_t* cat34  = (half_t*)alloc(64 * PF * 2);
  half_t* sL     = (half_t*)alloc(32 * PF * 2);
  half_t* sAB    = (half_t*)alloc(32 * PF * 2);
  half_t* saL    = (half_t*)alloc(32 * PF * 2);
  half_t* saAB   = (half_t*)alloc(32 * PF * 2);
  half_t* cat56  = (half_t*)alloc(64 * PF * 2);
  half_t* cat78  = (half_t*)alloc(3  * PF * 2);
  half_t* buf3   = (half_t*)alloc(3  * PF * 2);
  half_t* outh   = (half_t*)alloc(3  * PF * 2);
  half_t* cat80  = (half_t*)alloc(80 * PF * 2);
  half_t* itb    = (half_t*)alloc((size_t)32 * N * PP * 2);
  half_t* qb     = (half_t*)alloc((size_t)48 * N * PP * 2);
  half_t* kbuf   = (half_t*)alloc((size_t)48 * N * PP * 2);
  half_t* koutb  = (half_t*)alloc((size_t)48 * N * PP * 2);
  float*  mdn    = (float*)alloc((size_t)N * PP * 4);
  float*  mdd    = (float*)alloc((size_t)N * PP * 4);
  float*  stdb   = (float*)alloc((size_t)N * 96 * 4);
  float*  gb     = (float*)alloc((size_t)N * 96 * 4);
  half_t* wpack  = (half_t*)alloc((size_t)9 * 96 * 96 * 2);

  // ---- conv dispatch (template specialization on KSZ/CB/COT) ----
  auto run_conv = [&](const CP& p, const half_t* in, int inC, int incoff,
                      half_t* out, int outC, int outcoff, int Hh, int Ww, int dil) {
    int cinP = (p.cin + 31) & ~31, coutP = (p.cout + 15) & ~15;
    int CBv = cinP / 32, COTv = coutP / 16;
    int total = p.k * p.k * coutP * cinP;
    pack_weights<<<(total + 255) / 256, 256, 0, stream>>>(p.w, wpack, p.cout, p.cin, p.k, coutP, cinP);
    dim3 grid(Ww / 16, Hh, N * COTv);
#define LC(K, B, T)                                                             \
  conv_wmma<K, B, T><<<grid, 32, 0, stream>>>(in, inC, incoff, p.cin, wpack,    \
      p.b, out, outC, outcoff, p.cout, Hh, Ww, dil)
    int key = (p.k == 3 ? 300 : 100) + CBv * 10 + COTv;
    switch (key) {
      case 311: LC(3,1,1); break;  case 312: LC(3,1,2); break;  case 313: LC(3,1,3); break;
      case 321: LC(3,2,1); break;  case 322: LC(3,2,2); break;  case 323: LC(3,2,3); break;
      case 331: LC(3,3,1); break;  case 332: LC(3,3,2); break;  case 333: LC(3,3,3); break;
      case 111: LC(1,1,1); break;  case 112: LC(1,1,2); break;  case 113: LC(1,1,3); break;
      case 121: LC(1,2,1); break;  case 122: LC(1,2,2); break;  case 123: LC(1,2,3); break;
      case 131: LC(1,3,1); break;  case 132: LC(1,3,2); break;  case 133: LC(1,3,3); break;
      default: break;
    }
#undef LC
  };

  auto run_bb = [&](const BBp& B, const half_t* in, int inC, int incoff,
                    half_t* out, int outC, int outcoff) {
    run_conv(B.c[0],  in,   inC, incoff, t1,   96, 0,  H, W, 1);
    run_conv(B.c[1],  in,   inC, incoff, t1,   96, 16, H, W, 4);
    run_conv(B.c[2],  in,   inC, incoff, t1,   96, 48, H, W, 16);
    run_conv(B.c[3],  t1,   96, 0,       fcat, 96, 0,  H, W, 1);   // x
    run_conv(B.c[4],  fcat, 96, 0,       t1,   96, 0,  H, W, 2);
    run_conv(B.c[5],  fcat, 96, 0,       t1,   96, 16, H, W, 8);
    run_conv(B.c[6],  fcat, 96, 0,       t1,   96, 48, H, W, 32);
    run_conv(B.c[7],  t1,   96, 0,       fcat, 96, 32, H, W, 1);   // y
    run_conv(B.c[8],  fcat, 96, 32,      t1,   96, 0,  H, W, 4);
    run_conv(B.c[9],  fcat, 96, 32,      t1,   96, 16, H, W, 16);
    run_conv(B.c[10], fcat, 96, 32,      t1,   96, 48, H, W, 64);
    run_conv(B.c[11], t1,   96, 0,       fcat, 96, 64, H, W, 1);   // z
    ca_std_k<<<N * 96, 256, 0, stream>>>(fcat, stdb);
    ca_fc_k<<<N, 128, 0, stream>>>(stdb, B.ca.w1, B.ca.b1, B.ca.w2, B.ca.b2, gb, 96, 12);
    int tot = 96 * (int)PF;
    ca_scale_k<65536><<<(tot + 255) / 256, 256, 0, stream>>>(fcat, gb, 96, tot);
    run_conv(B.c[12], fcat, 96, 0, t2, 32, 0, H, W, 1);
    run_conv(B.c[13], t2,   32, 0, out, outC, outcoff, H, W, 1);
  };

  auto run_sa = [&](const SAp& S, const half_t* in, int inC, int incoff,
                    half_t* out, int outC, int outcoff) {
    int tot32 = N * 32 * PP;
    nearest_f16_k<48,48,256,256><<<(tot32 + 255) / 256, 256, 0, stream>>>(
        in, inC, incoff, itb, 32, 0, 32, tot32);
    int totm = N * PP;
    nearest_mask_k<48,48,256,256><<<(totm + 255) / 256, 256, 0, stream>>>(maskgt, mdn, totm);
    nearest_mask_k<48,48,256,256><<<(totm + 255) / 256, 256, 0, stream>>>(mask_d, mdd, totm);
    run_conv(S.s,  itb, 32, 0, qb,   48, 0, DS, DS, 1);
    run_conv(S.ns, itb, 32, 0, kbuf, 48, 0, DS, DS, 1);
    spa_attn_k<<<dim3(PP, N), 256, 0, stream>>>(qb, kbuf, mdn, mdd, koutb, 48);
    int totc = N * 32 * HW;
    copy_ch_f16<65536><<<(totc + 255) / 256, 256, 0, stream>>>(in, inC, incoff,
                                                               cat80, 80, 0, 32, totc);
    int totu = N * 48 * HW;
    nearest_f16_k<256,256,48,48><<<(totu + 255) / 256, 256, 0, stream>>>(
        koutb, 48, 0, cat80, 80, 32, 48, totu);
    run_conv(S.fin, cat80, 80, 0, out, outC, outcoff, H, W, 1);
  };

  // ---- network ----
  int tot3 = N * 3 * HW, tot1 = N * 1 * HW;
  f32_to_f16_ch<65536><<<(tot3 + 255) / 256, 256, 0, stream>>>(lab, bufIn3, 3, 3, 0, tot3);
  run_conv(convst, bufIn3, 3, 0, im, 33, 0, H, W, 1);
  f32_to_f16_ch<65536><<<(tot1 + 255) / 256, 256, 0, stream>>>(maskgt, im, 1, 33, 32, tot1);

  run_bb(bb1, im, 33, 0, cat12, 64, 0);
  run_bb(bb2, im, 33, 0, cat12, 64, 32);
  run_conv(c12l,  cat12, 64, 0, bufL,  32, 0, H, W, 1);
  run_conv(c12ab, cat12, 64, 0, bufAB, 32, 0, H, W, 1);

  run_bb(bb3, bufL,  32, 0, cat34, 64, 0);
  run_bb(bb4, bufAB, 32, 0, cat34, 64, 32);
  run_conv(c34l,  cat34, 64, 0, sL,  32, 0, H, W, 1);
  run_sa(sa1, sL, 32, 0, saL, 32, 0);
  run_conv(c34ab, cat34, 64, 0, sAB, 32, 0, H, W, 1);
  run_sa(sa2, sAB, 32, 0, saAB, 32, 0);

  run_bb(bb5, saL,  32, 0, cat56, 64, 0);
  run_bb(bb6, saAB, 32, 0, cat56, 64, 32);
  run_conv(c56l,  cat56, 64, 0, sL,  32, 0, H, W, 1);
  run_sa(sa3, sL, 32, 0, saL, 32, 0);
  run_conv(c56ab, cat56, 64, 0, sAB, 32, 0, H, W, 1);
  run_sa(sa4, sAB, 32, 0, saAB, 32, 0);

  run_bb(bb7, saL,  32, 0, cat78, 3, 0);   // 1 channel
  run_bb(bb8, saAB, 32, 0, cat78, 3, 1);   // 2 channels

  add_lab_k<<<(tot3 + 255) / 256, 256, 0, stream>>>(cat78, lab, buf3, tot3);
  run_conv(convfin, buf3, 3, 0, outh, 3, 0, H, W, 1);
  f16_to_f32_k<<<(tot3 + 255) / 256, 256, 0, stream>>>(outh, (float*)d_out, tot3);
}
